// TransformerBlock_30889404793480
// MI455X (gfx1250) — matrix operand order repaired and checked by reading, not compiled, not run on hardware
//
#include <hip/hip_runtime.h>
#include <hip/hip_bf16.h>
#include <math.h>

// ---------------------------------------------------------------------------
// Types / constants
// ---------------------------------------------------------------------------
typedef __bf16 bf16;
typedef __attribute__((ext_vector_type(16))) __bf16 v16bf;
typedef __attribute__((ext_vector_type(8)))  __bf16 v8bf;
typedef __attribute__((ext_vector_type(8)))  float  v8f;

constexpr int Bsz = 2;
constexpr int S   = 2048;
constexpr int D   = 2048;
constexpr int F   = 8192;
constexpr int H   = 16;
constexpr int DK  = 128;           // head dim
constexpr int M   = Bsz * S;       // 4096 rows of activations

#define WMMA_BF16(a_, b_, c_) \
  __builtin_amdgcn_wmma_f32_16x16x32_bf16(false, (a_), false, (b_), (short)0, (c_), false, false)

__device__ static inline v8f zero_v8f() {
  v8f v;
#pragma unroll
  for (int i = 0; i < 8; i++) v[i] = 0.0f;
  return v;
}

// A-fragment (16x32 bf16, MxK): lane = M row (lane%16); halves 0..7 = K[koff..koff+7],
// halves 8..15 = K[16+koff..16+koff+7], koff = 0 (lanes 0-15) / 8 (lanes 16-31).
__device__ static inline v16bf load_a_frag(const bf16* __restrict__ base, int ld,
                                           int row0, int k0) {
  int lane = threadIdx.x & 31;
  int r    = lane & 15;
  int koff = (lane < 16) ? 0 : 8;
  const bf16* p = base + (size_t)(row0 + r) * ld + k0;
  v8bf lo = *(const v8bf*)(p + koff);
  v8bf hi = *(const v8bf*)(p + 16 + koff);
  v16bf o;
#pragma unroll
  for (int i = 0; i < 8; i++) { o[i] = lo[i]; o[8 + i] = hi[i]; }
  return o;
}

// B-fragment (32x16 bf16, KxN) loaded from B^T stored row-major as [N,K]:
// lane = N col (lane%16); halves 0..7 = K[koff..koff+7], halves 8..15 =
// K[16+koff..16+koff+7], koff = 0 (lanes 0-15) / 8 (lanes 16-31).
__device__ static inline v16bf load_b_frag(const bf16* __restrict__ baseT, int ld,
                                           int n0, int k0) {
  int lane = threadIdx.x & 31;
  int n    = lane & 15;
  int ks   = (lane < 16) ? 0 : 8;
  const bf16* p = baseT + (size_t)(n0 + n) * ld + k0 + ks;
  v8bf lo = *(const v8bf*)(p);
  v8bf hi = *(const v8bf*)(p + 16);
  v16bf o;
#pragma unroll
  for (int i = 0; i < 8; i++) { o[i] = lo[i]; o[8 + i] = hi[i]; }
  return o;
}

// Async global -> LDS copy of 16 bytes per lane (CDNA5 ISA).
// vdst = LDS byte address (low 32 bits of generic shared pointer), tracked by ASYNCcnt.
__device__ static inline void async_copy_b128(unsigned lds_off, const bf16* gptr) {
  asm volatile("global_load_async_to_lds_b128 %0, %1, off"
               :: "v"(lds_off), "v"(gptr)
               : "memory");
}
__device__ static inline void wait_asynccnt_0() {
  asm volatile("s_wait_asynccnt 0x0" ::: "memory");
}
__device__ static inline void wait_asynccnt_8() {
  asm volatile("s_wait_asynccnt 0x8" ::: "memory");
}

// ---------------------------------------------------------------------------
// Elementwise / normalization kernels
// ---------------------------------------------------------------------------
__global__ void f32_to_bf16_kernel(const float* __restrict__ in, bf16* __restrict__ out, int n) {
  int i = blockIdx.x * blockDim.x + threadIdx.x;
  if (i < n) out[i] = (bf16)in[i];
}

__global__ __launch_bounds__(256) void rmsnorm_bf16_kernel(const float* __restrict__ x,
                                                           const float* __restrict__ g,
                                                           bf16* __restrict__ out, int cols) {
  int row = blockIdx.x;
  const float* xr = x + (size_t)row * cols;
  float ss = 0.f;
  for (int c = threadIdx.x; c < cols; c += 256) { float v = xr[c]; ss += v * v; }
  __shared__ float red[256];
  red[threadIdx.x] = ss;
  __syncthreads();
#pragma unroll
  for (int s = 128; s > 0; s >>= 1) {
    if (threadIdx.x < s) red[threadIdx.x] += red[threadIdx.x + s];
    __syncthreads();
  }
  float inv = rsqrtf(red[0] / (float)cols + 1e-5f);
  for (int c = threadIdx.x; c < cols; c += 256)
    out[(size_t)row * cols + c] = (bf16)(xr[c] * inv * g[c]);
}

// q0/k0: [B,S,H*DK] bf16 -> RoPE -> qr/kr: [B,H,S,DK] bf16
__global__ void rope_kernel(const bf16* __restrict__ q0, const bf16* __restrict__ k0,
                            bf16* __restrict__ qr, bf16* __restrict__ kr) {
  int idx = blockIdx.x * blockDim.x + threadIdx.x;
  constexpr int HALF = DK / 2;
  if (idx >= Bsz * S * H * HALF) return;
  int i  = idx & (HALF - 1);
  int hi = (idx / HALF) % H;
  int si = (idx / (HALF * H)) % S;
  int bi = idx / (HALF * H * S);
  float freq = __expf(-(2.0f * (float)i / (float)DK) * 9.2103403719761836f); // ln(10000)
  float ang  = (float)si * freq;
  float cs = cosf(ang), sn = sinf(ang);
  size_t src = ((size_t)(bi * S + si)) * D + hi * DK + 2 * i;
  size_t dst = (((size_t)bi * H + hi) * S + si) * (size_t)DK + 2 * i;
  float q1 = (float)q0[src], q2 = (float)q0[src + 1];
  qr[dst]     = (bf16)(cs * q1 - sn * q2);
  qr[dst + 1] = (bf16)(sn * q1 + cs * q2);
  float k1 = (float)k0[src], k2 = (float)k0[src + 1];
  kr[dst]     = (bf16)(cs * k1 - sn * k2);
  kr[dst + 1] = (bf16)(sn * k1 + cs * k2);
}

// v0: [B,S,H*DK] bf16 -> vT: [B,H,DK,S] bf16
__global__ void transpose_v_kernel(const bf16* __restrict__ v0, bf16* __restrict__ vT) {
  int idx = blockIdx.x * blockDim.x + threadIdx.x;
  if (idx >= Bsz * H * DK * S) return;
  int si = idx & (S - 1);
  int di = (idx >> 11) & (DK - 1);
  int hi = (idx >> 18) & (H - 1);
  int bi = idx >> 22;
  vT[idx] = v0[((size_t)(bi * S + si)) * D + hi * DK + di];
}

__global__ void silu_mul_kernel(const bf16* __restrict__ A, const bf16* __restrict__ Bv,
                                bf16* __restrict__ G, int n) {
  int i = blockIdx.x * blockDim.x + threadIdx.x;
  if (i >= n) return;
  float a = (float)A[i], b = (float)Bv[i];
  G[i] = (bf16)(a * b / (1.0f + __expf(-a)));
}

// ---------------------------------------------------------------------------
// LDS-tiled GEMM: C[M,N] = A[M,K](bf16) * W[N,K]^T(bf16) (+ RES).
// Block = 8 waves -> 128M x 128N tile; K-step 64; double-buffered LDS tiles
// filled with async global->LDS copies (ASYNCcnt pipelined).
// Each wave computes 32x64 via 2x4 WMMA accumulators.
// EPI: 1 = f32 out + f32 residual, 2 = bf16 out.
// ---------------------------------------------------------------------------
template <int EPI>
__global__ __launch_bounds__(256) void gemm_bf16_kernel(
    const bf16* __restrict__ A, const bf16* __restrict__ W,
    const float* __restrict__ RES, float* __restrict__ Cf, bf16* __restrict__ Cb,
    int Mdim, int Ndim, int Kdim) {
  __shared__ __align__(16) bf16 Ash[2][128 * 64];  // 2 x 16KB
  __shared__ __align__(16) bf16 Wsh[2][128 * 64];  // 2 x 16KB

  int tid  = threadIdx.x;
  int lane = tid & 31;
  int nb   = Ndim >> 7;
  int mblk = (blockIdx.x / nb) << 7;
  int nblk = (blockIdx.x % nb) << 7;

  int wid = tid >> 5;          // 0..7
  int wm  = wid >> 1;          // 0..3 : 32-row band
  int wn2 = wid & 1;           // 0..1 : 64-col band

  v8f acc[2][4];
#pragma unroll
  for (int mi = 0; mi < 2; mi++)
#pragma unroll
    for (int t = 0; t < 4; t++) acc[mi][t] = zero_v8f();

  // Stage ks (64 K-columns) into buffer buf: 8 async b128 instructions per wave.
  auto stage = [&](int ks, int buf) {
    int k0 = ks << 6;
#pragma unroll
    for (int i = 0; i < 4; i++) {
      int s   = tid + 256 * i;      // 1024 segments of 16B per tile
      int row = s >> 3;
      int c8  = (s & 7) << 3;       // element offset within 64-col slab
      async_copy_b128((unsigned)(uintptr_t)&Ash[buf][row * 64 + c8],
                      A + (size_t)(mblk + row) * Kdim + k0 + c8);
      async_copy_b128((unsigned)(uintptr_t)&Wsh[buf][row * 64 + c8],
                      W + (size_t)(nblk + row) * Kdim + k0 + c8);
    }
  };

  int nk = Kdim >> 6;
  stage(0, 0);
  for (int ks = 0; ks < nk; ks++) {
    int cur = ks & 1;
    if (ks + 1 < nk) {
      stage(ks + 1, cur ^ 1);
      wait_asynccnt_8();   // async loads complete in order: current stage landed
    } else {
      wait_asynccnt_0();
    }
    __syncthreads();       // all waves' tile writes visible

    const bf16* Ab = &Ash[cur][0];
    const bf16* Wb = &Wsh[cur][0];
#pragma unroll
    for (int kc = 0; kc < 64; kc += 32) {
      v16bf a0 = load_a_frag(Ab, 64, wm * 32, kc);
      v16bf a1 = load_a_frag(Ab, 64, wm * 32 + 16, kc);
#pragma unroll
      for (int t = 0; t < 4; t++) {
        v16bf bw = load_b_frag(Wb, 64, wn2 * 64 + 16 * t, kc);
        acc[0][t] = WMMA_BF16(a0, bw, acc[0][t]);
        acc[1][t] = WMMA_BF16(a1, bw, acc[1][t]);
      }
    }
    __syncthreads();       // done reading before this buffer is refilled
  }

  int rbase = (lane < 16) ? 0 : 8;
  int cb    = lane & 15;
  int m0    = mblk + wm * 32;
  int n0    = nblk + wn2 * 64;
#pragma unroll
  for (int mi = 0; mi < 2; mi++) {
#pragma unroll
    for (int t = 0; t < 4; t++) {
#pragma unroll
      for (int r = 0; r < 8; r++) {
        size_t idx = (size_t)(m0 + 16 * mi + rbase + r) * Ndim + n0 + 16 * t + cb;
        float v = acc[mi][t][r];
        if constexpr (EPI == 1) {
          Cf[idx] = v + RES[idx];
        } else {
          Cb[idx] = (bf16)v;
        }
      }
    }
  }
}

// ---------------------------------------------------------------------------
// Flash attention: one wave handles 16 query rows of one (b,h).
// Q,K: [B,H,S,DK] bf16; Vt: [B,H,DK,S] bf16; O: [B,S,D] bf16.
// ---------------------------------------------------------------------------
__global__ __launch_bounds__(256) void attn_kernel(const bf16* __restrict__ Q,
                                                   const bf16* __restrict__ Kt,
                                                   const bf16* __restrict__ Vt,
                                                   bf16* __restrict__ O) {
  __shared__ __align__(16) bf16 psh[8][16 * 32];  // per-wave 16x32 prob tile
  int gid   = blockIdx.x * blockDim.x + threadIdx.x;
  int wave  = gid >> 5;
  int lane  = threadIdx.x & 31;
  int wslot = threadIdx.x >> 5;
  constexpr int QT = S / 16;
  if (wave >= Bsz * H * QT) return;
  int bh = wave / QT;
  int q0 = (wave % QT) * 16;
  int bi = bh / H, hi = bh % H;
  const bf16* Qbh = Q  + (size_t)bh * S * DK;
  const bf16* Kbh = Kt + (size_t)bh * S * DK;
  const bf16* Vbh = Vt + (size_t)bh * DK * S;

  v16bf qf[4];
#pragma unroll
  for (int t = 0; t < 4; t++) qf[t] = load_a_frag(Qbh, DK, q0, 32 * t);

  float mrow[8], lrow[8];
  v8f oacc[8];
#pragma unroll
  for (int r = 0; r < 8; r++) { mrow[r] = -INFINITY; lrow[r] = 0.f; }
#pragma unroll
  for (int t = 0; t < 8; t++) oacc[t] = zero_v8f();

  int rbase = (lane < 16) ? 0 : 8;
  int cb    = lane & 15;
  const float scl = 0.08838834764831845f;  // 1/sqrt(128)

  for (int jb = 0; jb <= q0 + 15; jb += 32) {
    v8f s0 = zero_v8f(), s1 = zero_v8f();
#pragma unroll
    for (int t = 0; t < 4; t++) {
      s0 = WMMA_BF16(qf[t], load_b_frag(Kbh, DK, jb, 32 * t), s0);
      s1 = WMMA_BF16(qf[t], load_b_frag(Kbh, DK, jb + 16, 32 * t), s1);
    }
#pragma unroll
    for (int r = 0; r < 8; r++) {
      int qrow = q0 + rbase + r;
      float v0 = (jb + cb      <= qrow) ? s0[r] * scl : -INFINITY;
      float v1 = (jb + 16 + cb <= qrow) ? s1[r] * scl : -INFINITY;
      float mx = fmaxf(v0, v1);
#pragma unroll
      for (int off = 1; off < 16; off <<= 1) mx = fmaxf(mx, __shfl_xor(mx, off, 32));
      float mn = fmaxf(mrow[r], mx);
      float al = __expf(mrow[r] - mn);
      float p0 = __expf(v0 - mn);
      float p1 = __expf(v1 - mn);
      float rs = p0 + p1;
#pragma unroll
      for (int off = 1; off < 16; off <<= 1) rs += __shfl_xor(rs, off, 32);
      lrow[r] = lrow[r] * al + rs;
      mrow[r] = mn;
#pragma unroll
      for (int t = 0; t < 8; t++) oacc[t][r] *= al;
      psh[wslot][(rbase + r) * 32 + cb]      = (bf16)p0;
      psh[wslot][(rbase + r) * 32 + 16 + cb] = (bf16)p1;
    }
    // Same-wave LDS round trip converts C-layout probs into an A-fragment.
    v16bf pf = load_a_frag(&psh[wslot][0], 32, 0, 0);
#pragma unroll
    for (int t = 0; t < 8; t++) {
      v16bf vb = load_b_frag(Vbh, S, 16 * t, jb);
      oacc[t] = WMMA_BF16(pf, vb, oacc[t]);
    }
  }

#pragma unroll
  for (int r = 0; r < 8; r++) {
    float inv = 1.0f / lrow[r];
    int srow  = q0 + rbase + r;
#pragma unroll
    for (int t = 0; t < 8; t++)
      O[((size_t)(bi * S + srow)) * D + hi * DK + 16 * t + cb] = (bf16)(oacc[t][r] * inv);
  }
}

// ---------------------------------------------------------------------------
// Launch
// ---------------------------------------------------------------------------
extern "C" void kernel_launch(void* const* d_in, const int* in_sizes, int n_in,
                              void* d_out, int out_size, void* d_ws, size_t ws_size,
                              hipStream_t stream) {
  const float* x   = (const float*)d_in[0];
  const float* Wq  = (const float*)d_in[1];
  const float* Wk  = (const float*)d_in[2];
  const float* Wv  = (const float*)d_in[3];
  const float* Wo  = (const float*)d_in[4];
  const float* w1  = (const float*)d_in[5];
  const float* w2  = (const float*)d_in[6];
  const float* w3  = (const float*)d_in[7];
  const float* g1  = (const float*)d_in[8];
  const float* g2  = (const float*)d_in[9];
  float* out = (float*)d_out;

  char* ws = (char*)d_ws;
  size_t off = 0;
  auto alloc = [&](size_t bytes) -> char* {
    char* p = ws + off;
    off += (bytes + 255) & ~(size_t)255;
    return p;
  };

  bf16* Wq_b = (bf16*)alloc((size_t)D * D * 2);
  bf16* Wk_b = (bf16*)alloc((size_t)D * D * 2);
  bf16* Wv_b = (bf16*)alloc((size_t)D * D * 2);
  bf16* Wo_b = (bf16*)alloc((size_t)D * D * 2);
  bf16* w1_b = (bf16*)alloc((size_t)F * D * 2);
  bf16* w3_b = (bf16*)alloc((size_t)F * D * 2);
  bf16* w2_b = (bf16*)alloc((size_t)F * D * 2);
  bf16* h_b  = (bf16*)alloc((size_t)M * D * 2);
  float* xa  = (float*)alloc((size_t)M * D * 4);
  char* scratch = ws + off;  // overlay region: max(attention phase, FFN phase)

  // Phase 1 (attention) overlays
  bf16* q0 = (bf16*)scratch;
  bf16* k0 = q0 + (size_t)M * D;
  bf16* v0 = k0 + (size_t)M * D;
  bf16* qr = v0 + (size_t)M * D;
  bf16* kr = qr + (size_t)M * D;
  bf16* vT = kr + (size_t)M * D;
  bf16* ob = vT + (size_t)M * D;
  // Phase 2 (FFN) overlays — attention scratch is dead by then
  bf16* a_b = (bf16*)scratch;
  bf16* b_b = a_b + (size_t)M * F;
  bf16* g_b = b_b + (size_t)M * F;

  auto cvt = [&](const float* src, bf16* dst, size_t n) {
    f32_to_bf16_kernel<<<(int)((n + 255) / 256), 256, 0, stream>>>(src, dst, (int)n);
  };
  cvt(Wq, Wq_b, (size_t)D * D);
  cvt(Wk, Wk_b, (size_t)D * D);
  cvt(Wv, Wv_b, (size_t)D * D);
  cvt(Wo, Wo_b, (size_t)D * D);
  cvt(w1, w1_b, (size_t)F * D);
  cvt(w3, w3_b, (size_t)F * D);
  cvt(w2, w2_b, (size_t)F * D);

  rmsnorm_bf16_kernel<<<M, 256, 0, stream>>>(x, g1, h_b, D);

  int blkD = (M / 128) * (D / 128);   // 512 blocks
  gemm_bf16_kernel<2><<<blkD, 256, 0, stream>>>(h_b, Wq_b, nullptr, nullptr, q0, M, D, D);
  gemm_bf16_kernel<2><<<blkD, 256, 0, stream>>>(h_b, Wk_b, nullptr, nullptr, k0, M, D, D);
  gemm_bf16_kernel<2><<<blkD, 256, 0, stream>>>(h_b, Wv_b, nullptr, nullptr, v0, M, D, D);

  int tp = Bsz * S * H * (DK / 2);
  rope_kernel<<<(tp + 255) / 256, 256, 0, stream>>>(q0, k0, qr, kr);
  int tv = Bsz * H * DK * S;
  transpose_v_kernel<<<(tv + 255) / 256, 256, 0, stream>>>(v0, vT);

  int wavesA = Bsz * H * (S / 16);
  attn_kernel<<<(wavesA * 32 + 255) / 256, 256, 0, stream>>>(qr, kr, vT, ob);

  gemm_bf16_kernel<1><<<blkD, 256, 0, stream>>>(ob, Wo_b, x, xa, nullptr, M, D, D);

  rmsnorm_bf16_kernel<<<M, 256, 0, stream>>>(xa, g2, h_b, D);

  int blkF = (M / 128) * (F / 128);   // 2048 blocks
  gemm_bf16_kernel<2><<<blkF, 256, 0, stream>>>(h_b, w1_b, nullptr, nullptr, a_b, M, F, D);
  gemm_bf16_kernel<2><<<blkF, 256, 0, stream>>>(h_b, w3_b, nullptr, nullptr, b_b, M, F, D);

  int ng = M * F;
  silu_mul_kernel<<<(ng + 255) / 256, 256, 0, stream>>>(a_b, b_b, g_b, ng);

  gemm_bf16_kernel<1><<<blkD, 256, 0, stream>>>(g_b, w2_b, xa, out, nullptr, M, D, F);
}
